// RNNCellDRCNet_33380485825235
// MI455X (gfx1250) — matrix operand order repaired and checked by reading, not compiled, not run on hardware
//
#include <hip/hip_runtime.h>
#include <hip/hip_bf16.h>

#define SEQ    128
#define BATCH  1024
#define INDIM  256
#define HDIM   1024
#define PROBE1 33
#define PROBE2 95

typedef __attribute__((ext_vector_type(16))) __bf16        bf16x16;
typedef __attribute__((ext_vector_type(8)))  float         floatx8;
typedef __attribute__((ext_vector_type(4)))  unsigned int  uint32x4;

union Frag {
    bf16x16  v;
    uint32x4 q[2];
};

__device__ __forceinline__ unsigned short f32_to_bf16_rne(float f) {
    unsigned int u = __float_as_uint(f);
    u += 0x7FFFu + ((u >> 16) & 1u);        // round-to-nearest-even
    return (unsigned short)(u >> 16);
}

// ---------------- conversion / init kernels ----------------

__global__ void cvt_f32_to_bf16(const float* __restrict__ in,
                                unsigned short* __restrict__ out, int n) {
    int i = blockIdx.x * blockDim.x + threadIdx.x;
    if (i < n) out[i] = f32_to_bf16_rne(in[i]);
}

__global__ void zero_u16(unsigned short* __restrict__ p, int n) {
    int i = blockIdx.x * blockDim.x + threadIdx.x;
    if (i < n) p[i] = 0;
}

// ---------------- WMMA fragment loaders ----------------
// A-matrix 16x32 bf16 (ISA 7.12.2): lane L holds row (L&15);
// halves 0..7 = K kb..kb+7, halves 8..15 = K kb+16..kb+23, kb = k0 + ((L>=16)?8:0)
__device__ __forceinline__ Frag load_a_frag(const unsigned short* __restrict__ base,
                                            int ld, int row0, int k0, int lane) {
    const unsigned short* p = base + (size_t)(row0 + (lane & 15)) * ld
                                   + k0 + ((lane >> 4) << 3);
    Frag f;
    f.q[0] = *(const uint32x4*)(p);        // 8 halves (16 B)
    f.q[1] = *(const uint32x4*)(p + 16);   // 8 halves (16 B), +32 B from p
    return f;
}

// B-matrix 32x16 bf16: lane L holds column (L&15);
// halves 0..7 = K kb..kb+7, halves 8..15 = K kb+16..kb+23, kb = k0 + ((L>=16)?8:0)
// B[k,n] = W[n,k] with W row-major -> two 16-byte runs of W row n.
__device__ __forceinline__ Frag load_b_frag(const unsigned short* __restrict__ base,
                                            int ld, int col0, int k0, int lane) {
    const unsigned short* p = base + (size_t)(col0 + (lane & 15)) * ld
                                   + k0 + ((lane >> 4) << 3);
    Frag f;
    f.q[0] = *(const uint32x4*)(p);
    f.q[1] = *(const uint32x4*)(p + 16);
    return f;
}

// ---------------- one recurrent step: h_out = tanh(hx·W_hhT + x_t·W_ihT + b) ----------------
// Wave tile 32(M=batch) x 64(N=hidden). Block = 256 threads = 8 waves (2x4),
// block tile 64x256. Grid = (BATCH/64, HDIM/256) = (16, 4).
__global__ __launch_bounds__(256)
void rnn_step_wmma(const unsigned short* __restrict__ hx_in,   // [BATCH, HDIM] bf16
                   unsigned short*       __restrict__ hx_out,  // [BATCH, HDIM] bf16
                   const unsigned short* __restrict__ xt,      // [BATCH, INDIM] bf16
                   const unsigned short* __restrict__ Whh,     // [HDIM, HDIM]  bf16
                   const unsigned short* __restrict__ Wih,     // [HDIM, INDIM] bf16
                   const float* __restrict__ b_ih,
                   const float* __restrict__ b_hh,
                   float* __restrict__ probe)                  // null except t==PROBE1/2
{
    const int lane = threadIdx.x & 31;
    const int wave = threadIdx.x >> 5;
    const int m0 = blockIdx.x * 64  + (wave & 1) * 32;   // batch row base
    const int n0 = blockIdx.y * 256 + (wave >> 1) * 64;  // hidden col base

    floatx8 acc[2][4];
#pragma unroll
    for (int i = 0; i < 2; ++i)
#pragma unroll
        for (int j = 0; j < 4; ++j) acc[i][j] = {};

    // ---- K in [0,1024): hx · W_hhT ----
    for (int k0 = 0; k0 < HDIM; k0 += 32) {
        Frag a[2], b[4];
#pragma unroll
        for (int i = 0; i < 2; ++i) a[i] = load_a_frag(hx_in, HDIM, m0 + 16 * i, k0, lane);
#pragma unroll
        for (int j = 0; j < 4; ++j) b[j] = load_b_frag(Whh, HDIM, n0 + 16 * j, k0, lane);
#pragma unroll
        for (int i = 0; i < 2; ++i)
#pragma unroll
            for (int j = 0; j < 4; ++j)
                acc[i][j] = __builtin_amdgcn_wmma_f32_16x16x32_bf16(
                    false, a[i].v, false, b[j].v, (short)0, acc[i][j], false, false);
    }

    // ---- K in [0,256): x_t · W_ihT (fused input GEMM) ----
    for (int k0 = 0; k0 < INDIM; k0 += 32) {
        Frag a[2], b[4];
#pragma unroll
        for (int i = 0; i < 2; ++i) a[i] = load_a_frag(xt, INDIM, m0 + 16 * i, k0, lane);
#pragma unroll
        for (int j = 0; j < 4; ++j) b[j] = load_b_frag(Wih, INDIM, n0 + 16 * j, k0, lane);
#pragma unroll
        for (int i = 0; i < 2; ++i)
#pragma unroll
            for (int j = 0; j < 4; ++j)
                acc[i][j] = __builtin_amdgcn_wmma_f32_16x16x32_bf16(
                    false, a[i].v, false, b[j].v, (short)0, acc[i][j], false, false);
    }

    // ---- epilogue: bias + tanh, write bf16 hx (and fp32 probe snapshot) ----
    // C/D layout: VGPR v -> row = 16*i + v + ((lane>=16)?8:0), col = 16*j + (lane&15)
    const int rofs = (lane >> 4) << 3;
#pragma unroll
    for (int j = 0; j < 4; ++j) {
        const int col = n0 + 16 * j + (lane & 15);
        const float bias = b_ih[col] + b_hh[col];
#pragma unroll
        for (int i = 0; i < 2; ++i) {
#pragma unroll
            for (int v = 0; v < 8; ++v) {
                const int row = m0 + 16 * i + rofs + v;
                const float h = tanhf(acc[i][j][v] + bias);
                hx_out[(size_t)row * HDIM + col] = f32_to_bf16_rne(h);
                if (probe) probe[(size_t)row * HDIM + col] = h;
            }
        }
    }
}

// ---------------- final FC + sigmoid: out[w,b] = sigmoid(h_w[b]·W_fc + b_fc) ----------------
__global__ __launch_bounds__(256)
void fc_sigmoid(const float* __restrict__ h1, const float* __restrict__ h2,
                const float* __restrict__ Wfc, const float* __restrict__ bfc,
                float* __restrict__ out)
{
    const int gid  = blockIdx.x * blockDim.x + threadIdx.x;
    const int w    = gid >> 5;           // wave index 0..2047
    const int lane = gid & 31;
    const int which = w >> 10;           // 0 or 1
    const int bidx  = w & 1023;
    const float* h = which ? h2 : h1;

    float s = 0.f;
    for (int i = lane; i < HDIM; i += 32)
        s += h[(size_t)bidx * HDIM + i] * Wfc[i];
#pragma unroll
    for (int off = 16; off > 0; off >>= 1)
        s += __shfl_xor(s, off, 32);
    if (lane == 0)
        out[which * BATCH + bidx] = 1.f / (1.f + expf(-(s + bfc[0])));
}

// ---------------- host launcher ----------------
extern "C" void kernel_launch(void* const* d_in, const int* in_sizes, int n_in,
                              void* d_out, int out_size, void* d_ws, size_t ws_size,
                              hipStream_t stream) {
    const float* x    = (const float*)d_in[0];  // [SEQ, BATCH, INDIM]
    const float* W_ih = (const float*)d_in[1];  // [HDIM, INDIM]
    const float* W_hh = (const float*)d_in[2];  // [HDIM, HDIM]
    const float* b_ih = (const float*)d_in[3];  // [HDIM]
    const float* b_hh = (const float*)d_in[4];  // [HDIM]
    const float* W_fc = (const float*)d_in[5];  // [1, HDIM]
    const float* b_fc = (const float*)d_in[6];  // [1]
    float* out = (float*)d_out;                 // [2, BATCH]

    // workspace carve-out (~82 MB total)
    char* ws = (char*)d_ws;
    size_t off = 0;
    auto carve = [&](size_t bytes) {
        void* p = ws + off;
        off = (off + bytes + 255) & ~(size_t)255;
        return p;
    };
    unsigned short* xb   = (unsigned short*)carve((size_t)SEQ * BATCH * INDIM * 2);
    unsigned short* wihb = (unsigned short*)carve((size_t)HDIM * INDIM * 2);
    unsigned short* whhb = (unsigned short*)carve((size_t)HDIM * HDIM * 2);
    unsigned short* hxA  = (unsigned short*)carve((size_t)BATCH * HDIM * 2);
    unsigned short* hxB  = (unsigned short*)carve((size_t)BATCH * HDIM * 2);
    float* h1 = (float*)carve((size_t)BATCH * HDIM * 4);
    float* h2 = (float*)carve((size_t)BATCH * HDIM * 4);

    int n;
    n = SEQ * BATCH * INDIM;
    cvt_f32_to_bf16<<<(n + 255) / 256, 256, 0, stream>>>(x, xb, n);
    n = HDIM * INDIM;
    cvt_f32_to_bf16<<<(n + 255) / 256, 256, 0, stream>>>(W_ih, wihb, n);
    n = HDIM * HDIM;
    cvt_f32_to_bf16<<<(n + 255) / 256, 256, 0, stream>>>(W_hh, whhb, n);
    n = BATCH * HDIM;
    zero_u16<<<(n + 255) / 256, 256, 0, stream>>>(hxA, n);

    dim3 grid(BATCH / 64, HDIM / 256);  // (16, 4) blocks, 8 waves each
    for (int t = 0; t < SEQ; ++t) {
        const unsigned short* hin  = (t & 1) ? hxB : hxA;
        unsigned short*       hout = (t & 1) ? hxA : hxB;
        float* probe = (t == PROBE1) ? h1 : (t == PROBE2) ? h2 : nullptr;
        rnn_step_wmma<<<grid, 256, 0, stream>>>(
            hin, hout, xb + (size_t)t * BATCH * INDIM, whhb, wihb, b_ih, b_hh, probe);
    }

    // 2048 output scalars, one wave each -> 2048 waves = 256 blocks of 256 threads
    fc_sigmoid<<<256, 256, 0, stream>>>(h1, h2, W_fc, b_fc, out);
}
